// DeepseekV3FlashAttention2_73804718014885
// MI455X (gfx1250) — matrix operand order repaired and checked by reading, not compiled, not run on hardware
//
#include <hip/hip_runtime.h>
#include <math.h>

// ---------------- problem constants ----------------
constexpr int kS    = 2048;
constexpr int kHID  = 2048;
constexpr int kQLR  = 1536;
constexpr int kNH   = 16;
constexpr int kQHD  = 192;   // NOPE + ROPE
constexpr int kNOPE = 128;
constexpr int kROPE = 64;
constexpr int kKVLR = 512;
constexpr int kVD   = 128;
constexpr int kNHQHD = kNH * kQHD;          // 3072
constexpr int kNHKV  = kNH * (kNOPE + kVD); // 4096
constexpr int kCKVW  = kKVLR + kROPE;       // 576

// ---------------- vector types ----------------
typedef __attribute__((ext_vector_type(16))) __bf16 v16bf;
typedef __attribute__((ext_vector_type(8)))  __bf16 v8bf;
typedef __attribute__((ext_vector_type(8)))  float  v8f;

__device__ __forceinline__ v8f vzero8() {
  v8f x;
#pragma unroll
  for (int i = 0; i < 8; ++i) x[i] = 0.0f;
  return x;
}

// A-matrix fragment (16x32 bf16, M rows x K): per CDNA5 ISA 7.12.2
// lane<16: row=lane, K chunks {0..7} and {16..23}; lane>=16: row=lane-16, {8..15},{24..31}
__device__ __forceinline__ v16bf load_frag_a(const __bf16* base, int stride, int lane) {
  const __bf16* p = base + (size_t)(lane & 15) * stride + ((lane >> 4) << 3);
  v8bf lo = *(const v8bf*)p;
  v8bf hi = *(const v8bf*)(p + 16);
  v16bf r;
#pragma unroll
  for (int i = 0; i < 8; ++i) { r[i] = lo[i]; r[8 + i] = hi[i]; }
  return r;
}

// B-matrix fragment (32x16 bf16, K x N), source stored transposed (N rows x K):
// lane<16: col=lane, K chunks {0..7} and {16..23}; lane>=16: col=lane-16, {8..15},{24..31}
__device__ __forceinline__ v16bf load_frag_b(const __bf16* base, int stride, int lane) {
  const __bf16* p = base + (size_t)(lane & 15) * stride + ((lane >> 4) << 3);
  v8bf lo = *(const v8bf*)p;
  v8bf hi = *(const v8bf*)(p + 16);
  v16bf r;
#pragma unroll
  for (int i = 0; i < 8; ++i) { r[i] = lo[i]; r[8 + i] = hi[i]; }
  return r;
}

// ---------------- f32 -> bf16 flat convert ----------------
__global__ void convert_bf16_kernel(const float* __restrict__ in, __bf16* __restrict__ out, int n) {
  int i = blockIdx.x * blockDim.x + threadIdx.x;
  if (i < n) out[i] = (__bf16)in[i];
}

// ---------------- f32 (RxC) -> bf16 transposed (CxR) ----------------
__global__ __launch_bounds__(1024)
void transpose_bf16_kernel(const float* __restrict__ in, __bf16* __restrict__ out, int R, int C) {
  __shared__ float tile[32][33];
  int tx = threadIdx.x, ty = threadIdx.y;
  int c = blockIdx.x * 32 + tx, r = blockIdx.y * 32 + ty;
  if (r < R && c < C) tile[ty][tx] = in[(size_t)r * C + c];
  __syncthreads();
  int orow = blockIdx.x * 32 + ty;  // output row = original col
  int ocol = blockIdx.y * 32 + tx;  // output col = original row
  if (orow < C && ocol < R) out[(size_t)orow * R + ocol] = (__bf16)tile[tx][ty];
}

// ---------------- WMMA GEMM: C(MxN,f32) = A(MxK,bf16) * Bt(NxK,bf16)^T ----------------
// K-loop unrolled x2 with ping-pong fragment buffers: buffer loads always target the
// same variables (no rotation copies), and each 8-WMMA burst overlaps the other
// buffer's 12 in-flight global_load_b128s. Requires K % 64 == 0 (true for all calls).
__global__ __launch_bounds__(256)
void gemm_bf16_kernel(const __bf16* __restrict__ A, const __bf16* __restrict__ Bt,
                      float* __restrict__ C, int M, int N, int K) {
  int lane = threadIdx.x & 31;
  int wave = blockIdx.x * (blockDim.x >> 5) + (threadIdx.x >> 5);
  int ntiles = N >> 6;               // 64-wide N tiles per wave
  int mt = wave / ntiles, nt = wave % ntiles;
  if (mt >= (M >> 5)) return;        // wave-uniform exit; EXEC stays full for WMMA
  int m0 = mt * 32, n0 = nt * 64;

  const __bf16* A0 = A + (size_t)m0 * K;
  const __bf16* A1 = A + (size_t)(m0 + 16) * K;
  const __bf16* B0 = Bt + (size_t)n0 * K;
  const __bf16* B1 = B0 + (size_t)16 * K;
  const __bf16* B2 = B0 + (size_t)32 * K;
  const __bf16* B3 = B0 + (size_t)48 * K;

  v8f acc[2][4];
#pragma unroll
  for (int i = 0; i < 2; ++i)
#pragma unroll
    for (int j = 0; j < 4; ++j) acc[i][j] = vzero8();

  // stage X: fragments for kb; stage Y: fragments for kb+32
  v16bf xa0 = load_frag_a(A0, K, lane);
  v16bf xa1 = load_frag_a(A1, K, lane);
  v16bf xb0 = load_frag_b(B0, K, lane);
  v16bf xb1 = load_frag_b(B1, K, lane);
  v16bf xb2 = load_frag_b(B2, K, lane);
  v16bf xb3 = load_frag_b(B3, K, lane);

  for (int kb = 0; kb < K; kb += 64) {
    // issue stage-Y loads (kb+32, always in range)
    int ky = kb + 32;
    v16bf ya0 = load_frag_a(A0 + ky, K, lane);
    v16bf ya1 = load_frag_a(A1 + ky, K, lane);
    v16bf yb0 = load_frag_b(B0 + ky, K, lane);
    v16bf yb1 = load_frag_b(B1 + ky, K, lane);
    v16bf yb2 = load_frag_b(B2 + ky, K, lane);
    v16bf yb3 = load_frag_b(B3 + ky, K, lane);
    __builtin_prefetch(A0 + ky + 96, 0, 1);
    __builtin_prefetch(B0 + ky + 96, 0, 1);

    // compute on stage X
    acc[0][0] = __builtin_amdgcn_wmma_f32_16x16x32_bf16(false, xa0, false, xb0, (short)0, acc[0][0], false, false);
    acc[1][0] = __builtin_amdgcn_wmma_f32_16x16x32_bf16(false, xa1, false, xb0, (short)0, acc[1][0], false, false);
    acc[0][1] = __builtin_amdgcn_wmma_f32_16x16x32_bf16(false, xa0, false, xb1, (short)0, acc[0][1], false, false);
    acc[1][1] = __builtin_amdgcn_wmma_f32_16x16x32_bf16(false, xa1, false, xb1, (short)0, acc[1][1], false, false);
    acc[0][2] = __builtin_amdgcn_wmma_f32_16x16x32_bf16(false, xa0, false, xb2, (short)0, acc[0][2], false, false);
    acc[1][2] = __builtin_amdgcn_wmma_f32_16x16x32_bf16(false, xa1, false, xb2, (short)0, acc[1][2], false, false);
    acc[0][3] = __builtin_amdgcn_wmma_f32_16x16x32_bf16(false, xa0, false, xb3, (short)0, acc[0][3], false, false);
    acc[1][3] = __builtin_amdgcn_wmma_f32_16x16x32_bf16(false, xa1, false, xb3, (short)0, acc[1][3], false, false);

    // issue stage-X loads for kb+64 (clamped on last iteration: harmless reload of 0)
    int kx = (kb + 64 < K) ? (kb + 64) : 0;
    xa0 = load_frag_a(A0 + kx, K, lane);
    xa1 = load_frag_a(A1 + kx, K, lane);
    xb0 = load_frag_b(B0 + kx, K, lane);
    xb1 = load_frag_b(B1 + kx, K, lane);
    xb2 = load_frag_b(B2 + kx, K, lane);
    xb3 = load_frag_b(B3 + kx, K, lane);

    // compute on stage Y
    acc[0][0] = __builtin_amdgcn_wmma_f32_16x16x32_bf16(false, ya0, false, yb0, (short)0, acc[0][0], false, false);
    acc[1][0] = __builtin_amdgcn_wmma_f32_16x16x32_bf16(false, ya1, false, yb0, (short)0, acc[1][0], false, false);
    acc[0][1] = __builtin_amdgcn_wmma_f32_16x16x32_bf16(false, ya0, false, yb1, (short)0, acc[0][1], false, false);
    acc[1][1] = __builtin_amdgcn_wmma_f32_16x16x32_bf16(false, ya1, false, yb1, (short)0, acc[1][1], false, false);
    acc[0][2] = __builtin_amdgcn_wmma_f32_16x16x32_bf16(false, ya0, false, yb2, (short)0, acc[0][2], false, false);
    acc[1][2] = __builtin_amdgcn_wmma_f32_16x16x32_bf16(false, ya1, false, yb2, (short)0, acc[1][2], false, false);
    acc[0][3] = __builtin_amdgcn_wmma_f32_16x16x32_bf16(false, ya0, false, yb3, (short)0, acc[0][3], false, false);
    acc[1][3] = __builtin_amdgcn_wmma_f32_16x16x32_bf16(false, ya1, false, yb3, (short)0, acc[1][3], false, false);
  }

  int half = lane >> 4, col = lane & 15;
#pragma unroll
  for (int i = 0; i < 2; ++i)
#pragma unroll
    for (int j = 0; j < 4; ++j)
#pragma unroll
      for (int r = 0; r < 8; ++r)
        C[(size_t)(m0 + i * 16 + r + 8 * half) * N + (n0 + j * 16 + col)] = acc[i][j][r];
}

// ---------------- RMS-norm row kernel: f32 in -> bf16 out ----------------
__global__ __launch_bounds__(256)
void rmsnorm_bf16_kernel(const float* __restrict__ X, const float* __restrict__ w,
                         __bf16* __restrict__ Y, int width, int stride) {
  int row = blockIdx.x;
  const float* x = X + (size_t)row * stride;
  __shared__ float red[256];
  float ss = 0.0f;
  for (int j = threadIdx.x; j < width; j += 256) { float v = x[j]; ss += v * v; }
  red[threadIdx.x] = ss;
  __syncthreads();
  for (int s = 128; s > 0; s >>= 1) {
    if (threadIdx.x < (unsigned)s) red[threadIdx.x] += red[threadIdx.x + s];
    __syncthreads();
  }
  float inv = rsqrtf(red[0] / (float)width + 1e-6f);
  __bf16* y = Y + (size_t)row * width;
  for (int j = threadIdx.x; j < width; j += 256) y[j] = (__bf16)(w[j] * x[j] * inv);
}

// ---------------- RoPE + head pack: build Qh(NH,S,192), Kh(NH,S,192), Vt(NH,128,S) ----------------
__global__ __launch_bounds__(192)
void rope_pack_kernel(const float* __restrict__ Qf, const float* __restrict__ KVf,
                      const float* __restrict__ CKV, const int* __restrict__ pos,
                      __bf16* __restrict__ Qh, __bf16* __restrict__ Kh, __bf16* __restrict__ Vt) {
  int s = blockIdx.x, h = blockIdx.y, j = threadIdx.x;
  const float* qrow  = Qf  + (size_t)s * kNHQHD + h * kQHD;
  const float* kvrow = KVf + (size_t)s * kNHKV  + h * (kNOPE + kVD);
  size_t qo = ((size_t)h * kS + s) * kQHD;
  if (j < kNOPE) {
    Qh[qo + j] = (__bf16)qrow[j];
    Kh[qo + j] = (__bf16)kvrow[j];
    Vt[(size_t)h * kVD * kS + (size_t)j * kS + s] = (__bf16)kvrow[kNOPE + j];
  } else {
    int i = j - kNOPE;          // 0..63
    int f = i & 31;             // frequency index
    float t = (float)pos[s];
    float invf = exp2f(-(float)f * (13.287712379549449f / 32.0f)); // 10000^{-2f/64}
    float ang = t * invf;
    float c = cosf(ang), sn = sinf(ang);
    // deinterleave + rotate-half combine (matches reference exactly)
    const float* qpe = qrow + kNOPE;
    float x0 = qpe[2 * f], x1 = qpe[2 * f + 1];
    float qv = (i < 32) ? (x0 * c - x1 * sn) : (x1 * c + x0 * sn);
    Qh[qo + kNOPE + i] = (__bf16)qv;
    const float* kpe = CKV + (size_t)s * kCKVW + kKVLR;
    float y0 = kpe[2 * f], y1 = kpe[2 * f + 1];
    float kv = (i < 32) ? (y0 * c - y1 * sn) : (y1 * c + y0 * sn);
    Kh[qo + kNOPE + i] = (__bf16)kv;
  }
}

// ---------------- causal flash attention, one wave per (head, 16-row q tile) ----------------
__global__ __launch_bounds__(128)
void attn_kernel(const __bf16* __restrict__ Qh, const __bf16* __restrict__ Kh,
                 const __bf16* __restrict__ Vt, __bf16* __restrict__ O) {
  __shared__ __attribute__((aligned(16))) __bf16 lds[4][16 * 32];
  int lane = threadIdx.x & 31;
  int warp = threadIdx.x >> 5;
  int h  = blockIdx.y;
  int qt = blockIdx.x * 4 + warp;
  int q0 = qt * 16;
  const __bf16* Q  = Qh + (size_t)h * kS * kQHD;
  const __bf16* Kp = Kh + (size_t)h * kS * kQHD;
  const __bf16* Vp = Vt + (size_t)h * kVD * kS;
  __bf16* P = lds[warp];
  const float scale = 0.07216878364870322f; // 192^-0.5
  int half = lane >> 4, col = lane & 15;

  v16bf qf[6];
#pragma unroll
  for (int c = 0; c < 6; ++c)
    qf[c] = load_frag_a(Q + (size_t)q0 * kQHD + c * 32, kQHD, lane);

  v8f m, l;
  v8f o[8];
#pragma unroll
  for (int r = 0; r < 8; ++r) { m[r] = -1e30f; l[r] = 0.0f; }
#pragma unroll
  for (int t = 0; t < 8; ++t) o[t] = vzero8();

  int nkb = (q0 + 16 + 31) >> 5; // causal: keys 0 .. q0+15
  for (int kbi = 0; kbi < nkb; ++kbi) {
    int kb = kbi * 32;
    v8f s0 = vzero8(), s1 = vzero8();

    // scores: 12 WMMAs, K-fragments double-buffered ahead of each WMMA pair
    {
      v16bf b0 = load_frag_b(Kp + (size_t)kb * kQHD,        kQHD, lane);
      v16bf b1 = load_frag_b(Kp + (size_t)(kb + 16) * kQHD, kQHD, lane);
#pragma unroll
      for (int c = 0; c < 6; ++c) {
        v16bf nb0 = b0, nb1 = b1;
        if (c < 5) {
          nb0 = load_frag_b(Kp + (size_t)kb * kQHD + (c + 1) * 32,        kQHD, lane);
          nb1 = load_frag_b(Kp + (size_t)(kb + 16) * kQHD + (c + 1) * 32, kQHD, lane);
        }
        s0 = __builtin_amdgcn_wmma_f32_16x16x32_bf16(false, qf[c], false, b0, (short)0, s0, false, false);
        s1 = __builtin_amdgcn_wmma_f32_16x16x32_bf16(false, qf[c], false, b1, (short)0, s1, false, false);
        b0 = nb0; b1 = nb1;
      }
    }
    // scale + causal mask
#pragma unroll
    for (int r = 0; r < 8; ++r) {
      int row = q0 + r + 8 * half;
      float v0 = s0[r] * scale;
      float v1 = s1[r] * scale;
      s0[r] = (kb + col      > row) ? -1e30f : v0;
      s1[r] = (kb + 16 + col > row) ? -1e30f : v1;
    }
    // online softmax: new row max (reduce over 16 lanes within each half)
    v8f mn;
#pragma unroll
    for (int r = 0; r < 8; ++r) mn[r] = fmaxf(m[r], fmaxf(s0[r], s1[r]));
#pragma unroll
    for (int mask = 1; mask <= 8; mask <<= 1)
#pragma unroll
      for (int r = 0; r < 8; ++r) mn[r] = fmaxf(mn[r], __shfl_xor(mn[r], mask, 32));
    v8f al;
#pragma unroll
    for (int r = 0; r < 8; ++r) { al[r] = __expf(m[r] - mn[r]); m[r] = mn[r]; }
#pragma unroll
    for (int r = 0; r < 8; ++r) { s0[r] = __expf(s0[r] - mn[r]); s1[r] = __expf(s1[r] - mn[r]); }
    v8f ls;
#pragma unroll
    for (int r = 0; r < 8; ++r) ls[r] = s0[r] + s1[r];
#pragma unroll
    for (int mask = 1; mask <= 8; mask <<= 1)
#pragma unroll
      for (int r = 0; r < 8; ++r) ls[r] += __shfl_xor(ls[r], mask, 32);
#pragma unroll
    for (int r = 0; r < 8; ++r) l[r] = l[r] * al[r] + ls[r];
#pragma unroll
    for (int t = 0; t < 8; ++t)
#pragma unroll
      for (int r = 0; r < 8; ++r) o[t][r] *= al[r];

    // stage P (C-layout -> row-major 16x32 bf16) in LDS, reload as A-fragment
#pragma unroll
    for (int r = 0; r < 8; ++r) {
      P[(r + 8 * half) * 32 + col]      = (__bf16)s0[r];
      P[(r + 8 * half) * 32 + 16 + col] = (__bf16)s1[r];
    }
    asm volatile("s_wait_dscnt 0" ::: "memory"); // LDS in-order within wave; fence vs reorder
    v16bf pf = load_frag_a(P, 32, lane);

    // P x V: 8 WMMAs, V-fragments double-buffered
    {
      v16bf vf = load_frag_b(Vp + kb, kS, lane);
#pragma unroll
      for (int t = 0; t < 8; ++t) {
        v16bf nvf = vf;
        if (t < 7) nvf = load_frag_b(Vp + (size_t)((t + 1) * 16) * kS + kb, kS, lane);
        o[t] = __builtin_amdgcn_wmma_f32_16x16x32_bf16(false, pf, false, vf, (short)0, o[t], false, false);
        vf = nvf;
      }
    }
    asm volatile("" ::: "memory");
  }

  // epilogue: normalize and write attn (S, NH*VD) bf16
#pragma unroll
  for (int t = 0; t < 8; ++t)
#pragma unroll
    for (int r = 0; r < 8; ++r) {
      float val = o[t][r] / l[r];
      O[(size_t)(q0 + r + 8 * half) * (kNH * kVD) + h * kVD + t * 16 + col] = (__bf16)val;
    }
}

// ---------------- host side ----------------
static inline void* ws_take(char*& p, size_t bytes) {
  void* r = (void*)p;
  p += (bytes + 255) & ~(size_t)255;
  return r;
}

static inline void gemm_launch(const void* A, const void* Bt, float* C,
                               int M, int N, int K, hipStream_t st) {
  int waves = (M / 32) * (N / 64);
  int blocks = (waves + 7) / 8;
  gemm_bf16_kernel<<<blocks, 256, 0, st>>>((const __bf16*)A, (const __bf16*)Bt, C, M, N, K);
}

extern "C" void kernel_launch(void* const* d_in, const int* in_sizes, int n_in,
                              void* d_out, int out_size, void* d_ws, size_t ws_size,
                              hipStream_t stream) {
  (void)in_sizes; (void)n_in; (void)out_size; (void)ws_size;
  const float* hs    = (const float*)d_in[0];
  const int*   pos   = (const int*)  d_in[1];
  const float* Wqa   = (const float*)d_in[2];
  const float* qa_ln = (const float*)d_in[3];
  const float* Wqb   = (const float*)d_in[4];
  const float* Wkva  = (const float*)d_in[5];
  const float* kv_ln = (const float*)d_in[6];
  const float* Wkvb  = (const float*)d_in[7];
  const float* Wo    = (const float*)d_in[8];

  char* p = (char*)d_ws;
  void* hsb   = ws_take(p, (size_t)kS * kHID * 2);
  void* WqaT  = ws_take(p, (size_t)kQLR * kHID * 2);
  void* WqbT  = ws_take(p, (size_t)kNHQHD * kQLR * 2);
  void* WkvaT = ws_take(p, (size_t)kCKVW * kHID * 2);
  void* WkvbT = ws_take(p, (size_t)kNHKV * kKVLR * 2);
  void* WoT   = ws_take(p, (size_t)kHID * (kNH * kVD) * 2);
  float* Xa   = (float*)ws_take(p, (size_t)kS * kQLR * 4);
  void*  XaN  = ws_take(p, (size_t)kS * kQLR * 2);
  float* Qf   = (float*)ws_take(p, (size_t)kS * kNHQHD * 4);
  float* CKV  = (float*)ws_take(p, (size_t)kS * kCKVW * 4);
  void*  CkvN = ws_take(p, (size_t)kS * kKVLR * 2);
  float* KVf  = (float*)ws_take(p, (size_t)kS * kNHKV * 4);
  void*  Qh   = ws_take(p, (size_t)kNH * kS * kQHD * 2);
  void*  Kh   = ws_take(p, (size_t)kNH * kS * kQHD * 2);
  void*  Vt   = ws_take(p, (size_t)kNH * kVD * kS * 2);
  void*  attnB= ws_take(p, (size_t)kS * (kNH * kVD) * 2);

  // 0) precision conversion / weight transposes (bf16, N-major for B-fragments)
  {
    int n = kS * kHID;
    convert_bf16_kernel<<<(n + 255) / 256, 256, 0, stream>>>(hs, (__bf16*)hsb, n);
  }
  dim3 tb(32, 32);
  transpose_bf16_kernel<<<dim3(kQLR  / 32, kHID / 32), tb, 0, stream>>>(Wqa,  (__bf16*)WqaT,  kHID, kQLR);
  transpose_bf16_kernel<<<dim3(kNHQHD/ 32, kQLR / 32), tb, 0, stream>>>(Wqb,  (__bf16*)WqbT,  kQLR, kNHQHD);
  transpose_bf16_kernel<<<dim3(kCKVW / 32, kHID / 32), tb, 0, stream>>>(Wkva, (__bf16*)WkvaT, kHID, kCKVW);
  transpose_bf16_kernel<<<dim3(kNHKV / 32, kKVLR/ 32), tb, 0, stream>>>(Wkvb, (__bf16*)WkvbT, kKVLR, kNHKV);
  transpose_bf16_kernel<<<dim3(kHID  / 32, (kNH*kVD) / 32), tb, 0, stream>>>(Wo, (__bf16*)WoT, kNH * kVD, kHID);

  // 1) q_a = hs @ Wqa ; rmsnorm -> bf16
  gemm_launch(hsb, WqaT, Xa, kS, kQLR, kHID, stream);
  rmsnorm_bf16_kernel<<<kS, 256, 0, stream>>>(Xa, qa_ln, (__bf16*)XaN, kQLR, kQLR);

  // 2) q = q_a_n @ Wqb
  gemm_launch(XaN, WqbT, Qf, kS, kNHQHD, kQLR, stream);

  // 3) ckv = hs @ Wkva ; rmsnorm(c_kv) -> bf16
  gemm_launch(hsb, WkvaT, CKV, kS, kCKVW, kHID, stream);
  rmsnorm_bf16_kernel<<<kS, 256, 0, stream>>>(CKV, kv_ln, (__bf16*)CkvN, kKVLR, kCKVW);

  // 4) kv = ckv_n @ Wkvb
  gemm_launch(CkvN, WkvbT, KVf, kS, kNHKV, kKVLR, stream);

  // 5) RoPE + head-major packing (Qh/Kh row-major per head, V transposed per head)
  rope_pack_kernel<<<dim3(kS, kNH), kQHD, 0, stream>>>(Qf, KVf, CKV, pos,
                                                       (__bf16*)Qh, (__bf16*)Kh, (__bf16*)Vt);

  // 6) causal flash attention (one wave per 16-row q tile per head)
  attn_kernel<<<dim3((kS / 16) / 4, kNH), 128, 0, stream>>>((const __bf16*)Qh, (const __bf16*)Kh,
                                                            (const __bf16*)Vt, (__bf16*)attnB);

  // 7) out = attn @ Wo  (f32 result straight to d_out)
  gemm_launch(attnB, WoT, (float*)d_out, kS, kHID, kNH * kVD, stream);
}
